// LowrankLearnableHash_54443005444297
// MI455X (gfx1250) — matrix operand order repaired and checked by reading, not compiled, not run on hardware
//
#include <hip/hip_runtime.h>
#include <hip/hip_bf16.h>

typedef __attribute__((ext_vector_type(16))) _Float16 v16h;
typedef __attribute__((ext_vector_type(8)))  _Float16 v8h;
typedef __attribute__((ext_vector_type(8)))  float    v8f;

#define RESg 192
#define RANK 8
#define NFPD 32
#define NI   128
#define NRAY 2048
#define NPTS (NRAY*NI)
#define RADIUS_F 1.3f
#define STEP_F   0.0352f

// ---------------- WMMA helpers ----------------

__device__ __forceinline__ v8f wmma16(v16h a, v16h b, v8f c) {
  return __builtin_amdgcn_wmma_f32_16x16x32_f16(
      /*neg_a=*/false, a, /*neg_b=*/false, b,
      /*c_mod=*/(short)0, c, /*reuse_a=*/false, /*reuse_b=*/false);
}

// A operand: 16x32 f16.  lanes0-15: row=lane, elems0-7=K0..7, elems8-15=K16..23
//            lanes16-31: row=lane-16, elems0-7=K8..15, elems8-15=K24..31
// Activations stored row-major f16 in LDS with stride 128 halves.
__device__ __forceinline__ v16h load_A(const _Float16* act, int lane, int kb) {
  int row  = lane & 15;
  int koff = (lane & 16) ? 8 : 0;
  const _Float16* base = act + row * 128 + kb + koff;
  v8h lo = *(const v8h*)(base);
  v8h hi = *(const v8h*)(base + 16);
  v16h a;
#pragma unroll
  for (int i = 0; i < 8; ++i) { a[i] = lo[i]; a[i + 8] = hi[i]; }
  return a;
}

// B operand: 32x16 f16. lane<16: N=lane, elems0-7=K0..7, elems8-15=K16..23; lane>=16: N=lane-16, K8..15 then K24..31.
// Weights pre-packed so each lane does one contiguous 32B load.
__device__ __forceinline__ v16h load_B(const _Float16* w, int tile, int lane) {
  return *(const v16h*)(w + tile * 512 + lane * 16);
}

// C/D: VGPR g -> M = g + (lane>=16 ? 8 : 0), N = lane&15
__device__ __forceinline__ void store_D_relu(v8f d, _Float16* act, int lane, int nb) {
  int col = nb + (lane & 15);
  int mb  = (lane & 16) ? 8 : 0;
#pragma unroll
  for (int g = 0; g < 8; ++g) {
    float v = d[g];
    v = v > 0.f ? v : 0.f;
    act[(g + mb) * 128 + col] = (_Float16)v;
  }
}

// ---------------- weight repack ----------------
// Pack row-major f32 weight [K x N] into f16 B-tiles. Tile (kt,nt) covers
// K in [kt*32,kt*32+32), N in [nt*16,nt*16+16). Element (lane,h):
//   n = nt*16 + (lane&15);  k = kt*32 + (h&7) + (lane&16 ? 8 : 0) + (h&8 ? 16 : 0)
__global__ void pack_w(const float* __restrict__ W, int K, int N, int Nt,
                       int total, _Float16* __restrict__ dst) {
  int idx = blockIdx.x * blockDim.x + threadIdx.x;
  if (idx >= total) return;
  int tile = idx >> 9, within = idx & 511;
  int lane = within >> 4, h = within & 15;
  int kt = tile / Nt, nt = tile % Nt;
  int n = nt * 16 + (lane & 15);
  int k = kt * 32 + (h & 7) + ((lane & 16) ? 8 : 0) + ((h & 8) ? 16 : 0);
  float v = (k < K && n < N) ? W[(size_t)k * N + n] : 0.f;
  dst[idx] = (_Float16)v;
}

// ---------------- misc ----------------

__device__ __forceinline__ void prep1(float c, int size, int& i0, float& w) {
  float idx = (c + 1.f) * 0.5f * (float)(size - 1);
  idx = fminf(fmaxf(idx, 0.f), (float)(size - 1));
  int i = (int)idx;                 // floor (idx >= 0)
  if (i > size - 2) i = size - 2;
  i0 = i;
  w  = idx - (float)i;
}

// ---------------- main per-sample kernel (1 wave / 16 points) ----------------

__global__ __launch_bounds__(32) void nerf_main(
    const float* __restrict__ rays_o, const float* __restrict__ rays_d,
    const float* __restrict__ Gxy, const float* __restrict__ Gxz,
    const float* __restrict__ Gyz, const float* __restrict__ Fg,
    const _Float16* __restrict__ wS0, const _Float16* __restrict__ wS1,
    const _Float16* __restrict__ wS2, const _Float16* __restrict__ wC0,
    const _Float16* __restrict__ wC1, const _Float16* __restrict__ wC2,
    float* __restrict__ sigma_ws, float* __restrict__ rgb_ws) {
  __shared__ __align__(32) _Float16 actA[16 * 128];
  __shared__ __align__(32) _Float16 actB[16 * 128];
  __shared__ _Float16 shb[16 * 16];
  __shared__ float maskb[16];

  const int lane = threadIdx.x;
  const int gid0 = blockIdx.x * 16;

  // ---- feature phase: lanes 0..15, one sample point each ----
  if (lane < 16) {
    int gid = gid0 + lane;
    int ray = gid >> 7;
    int smp = gid & (NI - 1);
    float rox = rays_o[ray * 3 + 0], roy = rays_o[ray * 3 + 1], roz = rays_o[ray * 3 + 2];
    float rdx = rays_d[ray * 3 + 0], rdy = rays_d[ray * 3 + 1], rdz = rays_d[ray * 3 + 2];

    float dx = fabsf(rdx) < 1e-9f ? 1e-9f : rdx;
    float dy = fabsf(rdy) < 1e-9f ? 1e-9f : rdy;
    float dz = fabsf(rdz) < 1e-9f ? 1e-9f : rdz;
    float t1x = (-RADIUS_F - rox) / dx, t2x = (RADIUS_F - rox) / dx;
    float t1y = (-RADIUS_F - roy) / dy, t2y = (RADIUS_F - roy) / dy;
    float t1z = (-RADIUS_F - roz) / dz, t2z = (RADIUS_F - roz) / dz;
    float tmin = fmaxf(fminf(t1x, t2x), fmaxf(fminf(t1y, t2y), fminf(t1z, t2z)));
    float tmax = fminf(fmaxf(t1x, t2x), fminf(fmaxf(t1y, t2y), fmaxf(t1z, t2z)));
    float nearv = fmaxf(tmin, 0.f);
    float tmid = nearv + ((float)smp + 0.5f) * STEP_F;
    float px = rox + tmid * rdx, py = roy + tmid * rdy, pz = roz + tmid * rdz;
    bool m = (fabsf(px) <= RADIUS_F) && (fabsf(py) <= RADIUS_F) &&
             (fabsf(pz) <= RADIUS_F) && (tmid <= tmax) && (tmax > tmin);
    maskb[lane] = m ? 1.f : 0.f;

    // SH basis of normalized direction (per-ray)
    {
      float invn = rsqrtf(rdx * rdx + rdy * rdy + rdz * rdz);
      float X = rdx * invn, Y = rdy * invn, Z = rdz * invn;
      float X2 = X * X, Y2 = Y * Y, Z2 = Z * Z;
      float sh[16];
      sh[0] = 0.28209479177387814f;
      sh[1] = -0.48860251190291987f * Y;
      sh[2] = 0.48860251190291987f * Z;
      sh[3] = -0.48860251190291987f * X;
      sh[4] = 1.0925484305920792f * X * Y;
      sh[5] = -1.0925484305920792f * Y * Z;
      sh[6] = 0.94617469575756f * Z2 - 0.31539156525252f;
      sh[7] = -1.0925484305920792f * X * Z;
      sh[8] = 0.5462742152960396f * (X2 - Y2);
      sh[9] = 0.5900435899266435f * Y * (-3.f * X2 + Y2);
      sh[10] = 2.890611442640554f * X * Y * Z;
      sh[11] = 0.4570457994644657f * Y * (1.f - 5.f * Z2);
      sh[12] = 0.3731763325901154f * Z * (5.f * Z2 - 3.f);
      sh[13] = 0.4570457994644657f * X * (1.f - 5.f * Z2);
      sh[14] = 1.445305721320277f * Z * (X2 - Y2);
      sh[15] = 0.5900435899266435f * X * (-X2 + 3.f * Y2);
#pragma unroll
      for (int j = 0; j < 16; ++j) shb[lane * 16 + j] = (_Float16)sh[j];
    }

    // CP-decomposed grid trilinear (no materialization): coords p = pts/RADIUS
    float cx = px / RADIUS_F, cy = py / RADIUS_F, cz = pz / RADIUS_F;
    int x0, y0, z0; float wx, wy, wz;
    prep1(cx, RESg, x0, wx);
    prep1(cy, RESg, y0, wy);
    prep1(cz, RESg, z0, wz);
    int xi[2] = {x0, x0 + 1}, yi[2] = {y0, y0 + 1}, zi[2] = {z0, z0 + 1};
    float wxv[2] = {1.f - wx, wx}, wyv[2] = {1.f - wy, wy}, wzv[2] = {1.f - wz, wz};
    float gv0 = 0.f, gv1 = 0.f, gv2 = 0.f;
    for (int a = 0; a < 2; ++a)
      for (int b = 0; b < 2; ++b)
        for (int c = 0; c < 2; ++c) {
          const float* pxy = Gxy + (size_t)(zi[a] * RESg + yi[b]) * 24; // [ch*8+r]
          const float* pxz = Gxz + (size_t)(zi[a] * RESg + xi[c]) * 24;
          const float* pyz = Gyz + (size_t)(yi[b] * RESg + xi[c]) * 24;
          float w = wzv[a] * wyv[b] * wxv[c];
          float s0 = 0.f, s1 = 0.f, s2 = 0.f;
#pragma unroll
          for (int r = 0; r < RANK; ++r) {
            s0 += pxy[r] * pxz[r] * pyz[r];
            s1 += pxy[8 + r] * pxz[8 + r] * pyz[8 + r];
            s2 += pxy[16 + r] * pxz[16 + r] * pyz[16 + r];
          }
          gv0 += w * s0; gv1 += w * s1; gv2 += w * s2;
        }

    // Feature-grid trilinear at Gvals
    int fx0, fy0, fz0; float fwx, fwy, fwz;
    prep1(gv0, NFPD, fx0, fwx);
    prep1(gv1, NFPD, fy0, fwy);
    prep1(gv2, NFPD, fz0, fwz);
    float Fv[32];
#pragma unroll
    for (int f = 0; f < 32; ++f) Fv[f] = 0.f;
    int fxi[2] = {fx0, fx0 + 1}, fyi[2] = {fy0, fy0 + 1}, fzi[2] = {fz0, fz0 + 1};
    float fwxv[2] = {1.f - fwx, fwx}, fwyv[2] = {1.f - fwy, fwy}, fwzv[2] = {1.f - fwz, fwz};
    for (int a = 0; a < 2; ++a)
      for (int b = 0; b < 2; ++b)
        for (int c = 0; c < 2; ++c) {
          const float* src = Fg + ((size_t)(fzi[a] * NFPD + fyi[b]) * NFPD + fxi[c]) * 32;
          float w = fwzv[a] * fwyv[b] * fwxv[c];
#pragma unroll
          for (int f = 0; f < 32; ++f) Fv[f] += w * src[f];
        }
#pragma unroll
    for (int f = 0; f < 32; ++f) actA[lane * 128 + f] = (_Float16)Fv[f];
  }
  __syncthreads();

  // ---- sigma MLP: 32 -> 128 -> 128 -> 16 ----
  {
    v16h aF = load_A(actA, lane, 0);
#pragma unroll
    for (int nt = 0; nt < 8; ++nt) {
      v8f acc = {};
      acc = wmma16(aF, load_B(wS0, nt, lane), acc);
      store_D_relu(acc, actB, lane, nt * 16);
    }
  }
  __syncthreads();
  {
    v16h a0 = load_A(actB, lane, 0);
    v16h a1 = load_A(actB, lane, 32);
    v16h a2 = load_A(actB, lane, 64);
    v16h a3 = load_A(actB, lane, 96);
#pragma unroll
    for (int nt = 0; nt < 8; ++nt) {
      v8f acc = {};
      acc = wmma16(a0, load_B(wS1, 0 * 8 + nt, lane), acc);
      acc = wmma16(a1, load_B(wS1, 1 * 8 + nt, lane), acc);
      acc = wmma16(a2, load_B(wS1, 2 * 8 + nt, lane), acc);
      acc = wmma16(a3, load_B(wS1, 3 * 8 + nt, lane), acc);
      store_D_relu(acc, actA, lane, nt * 16);
    }
  }
  __syncthreads();
  v8f accS = {};
  {
    v16h a0 = load_A(actA, lane, 0);
    v16h a1 = load_A(actA, lane, 32);
    v16h a2 = load_A(actA, lane, 64);
    v16h a3 = load_A(actA, lane, 96);
    accS = wmma16(a0, load_B(wS2, 0, lane), accS);
    accS = wmma16(a1, load_B(wS2, 1, lane), accS);
    accS = wmma16(a2, load_B(wS2, 2, lane), accS);
    accS = wmma16(a3, load_B(wS2, 3, lane), accS);
  }
  __syncthreads();

  // out16: col0 -> sigma (relu * mask, to global); cols1..15 -> ci cols 16..30
  {
    int col = lane & 15;
    int mb  = (lane & 16) ? 8 : 0;
#pragma unroll
    for (int g = 0; g < 8; ++g) {
      int M = g + mb;
      float v = accS[g];
      if (col == 0) {
        float s = v > 0.f ? v : 0.f;
        sigma_ws[gid0 + M] = s * maskb[M];
      } else {
        actA[M * 128 + col + 15] = (_Float16)v;
      }
    }
  }
  if (lane < 16) {
#pragma unroll
    for (int j = 0; j < 16; ++j) actA[lane * 128 + j] = shb[lane * 16 + j];
    actA[lane * 128 + 31] = (_Float16)0.f;
  }
  __syncthreads();

  // ---- color MLP: 31(pad32) -> 64 -> 64 -> 3(pad16) ----
  {
    v16h aci = load_A(actA, lane, 0);
#pragma unroll
    for (int nt = 0; nt < 4; ++nt) {
      v8f acc = {};
      acc = wmma16(aci, load_B(wC0, nt, lane), acc);
      store_D_relu(acc, actB, lane, nt * 16);
    }
  }
  __syncthreads();
  {
    v16h a0 = load_A(actB, lane, 0);
    v16h a1 = load_A(actB, lane, 32);
#pragma unroll
    for (int nt = 0; nt < 4; ++nt) {
      v8f acc = {};
      acc = wmma16(a0, load_B(wC1, 0 * 4 + nt, lane), acc);
      acc = wmma16(a1, load_B(wC1, 1 * 4 + nt, lane), acc);
      store_D_relu(acc, actA, lane, nt * 16);
    }
  }
  __syncthreads();
  {
    v16h a0 = load_A(actA, lane, 0);
    v16h a1 = load_A(actA, lane, 32);
    v8f acc = {};
    acc = wmma16(a0, load_B(wC2, 0, lane), acc);
    acc = wmma16(a1, load_B(wC2, 1, lane), acc);
    int col = lane & 15;
    int mb  = (lane & 16) ? 8 : 0;
    if (col < 3) {
#pragma unroll
      for (int g = 0; g < 8; ++g) {
        int M = g + mb;
        rgb_ws[(size_t)col * NPTS + gid0 + M] = acc[g] * maskb[M];
      }
    }
  }
}

// ---------------- per-ray compositing ----------------

__global__ void composite(const float* __restrict__ sigma,
                          const float* __restrict__ rgb,
                          float* __restrict__ out) {
  int r = blockIdx.x * blockDim.x + threadIdx.x;
  if (r >= NRAY) return;
  float T = 1.f, acc = 0.f, o0 = 0.f, o1 = 0.f, o2 = 0.f;
  for (int i = 0; i < NI; ++i) {
    int gid = r * NI + i;
    float s = sigma[gid];
    float a = 1.f - __expf(-s * STEP_F);
    float w = a * T;
    T *= (1.f - a + 1e-10f);
    acc += w;
    float c0 = rgb[gid];
    float c1 = rgb[NPTS + gid];
    float c2 = rgb[2 * NPTS + gid];
    o0 += w * (1.f / (1.f + __expf(-c0)));
    o1 += w * (1.f / (1.f + __expf(-c1)));
    o2 += w * (1.f / (1.f + __expf(-c2)));
  }
  float bg = 1.f - acc;
  out[r * 3 + 0] = o0 + bg;
  out[r * 3 + 1] = o1 + bg;
  out[r * 3 + 2] = o2 + bg;
}

// ---------------- launcher ----------------

extern "C" void kernel_launch(void* const* d_in, const int* in_sizes, int n_in,
                              void* d_out, int out_size, void* d_ws, size_t ws_size,
                              hipStream_t stream) {
  (void)in_sizes; (void)n_in; (void)out_size; (void)ws_size;
  const float* rays_o = (const float*)d_in[0];
  const float* rays_d = (const float*)d_in[1];
  const float* Gxy    = (const float*)d_in[2];
  const float* Gxz    = (const float*)d_in[3];
  const float* Gyz    = (const float*)d_in[4];
  const float* Fg     = (const float*)d_in[5];
  const float* sW0    = (const float*)d_in[6];
  const float* sW1    = (const float*)d_in[7];
  const float* sW2    = (const float*)d_in[8];
  const float* cW0    = (const float*)d_in[9];
  const float* cW1    = (const float*)d_in[10];
  const float* cW2    = (const float*)d_in[11];

  char* ws = (char*)d_ws;
  _Float16* wS0 = (_Float16*)ws;        // 8 tiles  (1 kt x 8 nt)
  _Float16* wS1 = wS0 + 8 * 512;        // 32 tiles (4 x 8)
  _Float16* wS2 = wS1 + 32 * 512;       // 4 tiles  (4 x 1)
  _Float16* wC0 = wS2 + 4 * 512;        // 4 tiles  (1 x 4)
  _Float16* wC1 = wC0 + 4 * 512;        // 8 tiles  (2 x 4)
  _Float16* wC2 = wC1 + 8 * 512;        // 2 tiles  (2 x 1)
  float* sigma_ws = (float*)(ws + 65536);
  float* rgb_ws   = sigma_ws + NPTS;    // 3 planes of NPTS

  pack_w<<<(8 * 512 + 255) / 256, 256, 0, stream>>>(sW0, 32, 128, 8, 8 * 512, wS0);
  pack_w<<<(32 * 512 + 255) / 256, 256, 0, stream>>>(sW1, 128, 128, 8, 32 * 512, wS1);
  pack_w<<<(4 * 512 + 255) / 256, 256, 0, stream>>>(sW2, 128, 16, 1, 4 * 512, wS2);
  pack_w<<<(4 * 512 + 255) / 256, 256, 0, stream>>>(cW0, 31, 64, 4, 4 * 512, wC0);
  pack_w<<<(8 * 512 + 255) / 256, 256, 0, stream>>>(cW1, 64, 64, 4, 8 * 512, wC1);
  pack_w<<<(2 * 512 + 255) / 256, 256, 0, stream>>>(cW2, 64, 3, 1, 2 * 512, wC2);

  nerf_main<<<NPTS / 16, 32, 0, stream>>>(rays_o, rays_d, Gxy, Gxz, Gyz, Fg,
                                          wS0, wS1, wS2, wC0, wC1, wC2,
                                          sigma_ws, rgb_ws);

  composite<<<(NRAY + 127) / 128, 128, 0, stream>>>(sigma_ws, rgb_ws, (float*)d_out);
}
